// ITNSR_2937757630599
// MI455X (gfx1250) — matrix operand order repaired and checked by reading, not compiled, not run on hardware
//
#include <hip/hip_runtime.h>
#include <hip/hip_bf16.h>
#include <stdint.h>

// ITSRN forward, fused for MI455X (gfx1250, wave32, WMMA f16->f32).
// Pipeline: enc conv -> pack weights f16 -> prep(layer1+score) ->
//           WMMA gemm2 (256x256) -> WMMA gemm3 fused with value-contraction ->
//           final weight MLP. wmat (113MB/shift) is never materialized.

#define B_     2
#define Q_     8192
#define H_     64
#define W_     64
#define C_     64
#define C9_    576
#define N3_    1728
#define HID_   256
#define NROWS_ 65536   // 4 shifts * B * Q

typedef __attribute__((ext_vector_type(16))) _Float16 v16h;
typedef __attribute__((ext_vector_type(8)))  float    v8f;

union AFrag { v16h v; uint32_t u[8]; };

__device__ __forceinline__ float gelu_tanh(float x) {
  float x3 = x * x * x;
  return 0.5f * x * (1.0f + tanhf(0.7978845608028654f * (x + 0.044715f * x3)));
}

// ---------------------------------------------------------------- encoder conv
__global__ __launch_bounds__(256) void itnsr_enc(
    const float* __restrict__ inp, const float* __restrict__ w,
    const float* __restrict__ bias, float* __restrict__ feat)
{
  int idx = blockIdx.x * 256 + threadIdx.x;          // B*C*H*W = 524288
  if (idx >= B_ * C_ * H_ * W_) return;
  int x = idx & 63, y = (idx >> 6) & 63, o = (idx >> 12) & 63, b = idx >> 18;
  float acc = bias[o];
  #pragma unroll
  for (int ci = 0; ci < 3; ++ci)
    #pragma unroll
    for (int dy = 0; dy < 3; ++dy)
      #pragma unroll
      for (int dx = 0; dx < 3; ++dx) {
        int yy = y + dy - 1, xx = x + dx - 1;
        if ((unsigned)yy < 64u && (unsigned)xx < 64u)
          acc += inp[(((size_t)b * 3 + ci) << 12) + (yy << 6) + xx] *
                 w[((o * 3 + ci) * 3 + dy) * 3 + dx];
      }
  feat[idx] = acc;
}

// ------------------------------------------- pack iw1/iw2 to f16, K-pair-packed
// Packed layout: p[(k>>1)*N + n] = {f16(W[k][n]), f16(W[k+1][n])} in one dword.
__global__ __launch_bounds__(256) void itnsr_packw(
    const float* __restrict__ iw1, const float* __restrict__ iw2,
    uint32_t* __restrict__ iw1p, uint32_t* __restrict__ iw2p)
{
  int idx = blockIdx.x * 256 + threadIdx.x;
  const int N1 = 128 * HID_;     // 32768 pairs
  const int N2 = 128 * N3_;      // 221184 pairs
  if (idx < N1) {
    int k2 = idx / HID_, n = idx - k2 * HID_;
    union { _Float16 h[2]; uint32_t u; } p;
    p.h[0] = (_Float16)iw1[(size_t)(2 * k2) * HID_ + n];
    p.h[1] = (_Float16)iw1[(size_t)(2 * k2 + 1) * HID_ + n];
    iw1p[idx] = p.u;
  } else if (idx < N1 + N2) {
    int t = idx - N1;
    int k2 = t / N3_, n = t - k2 * N3_;
    union { _Float16 h[2]; uint32_t u; } p;
    p.h[0] = (_Float16)iw2[(size_t)(2 * k2) * N3_ + n];
    p.h[1] = (_Float16)iw2[(size_t)(2 * k2 + 1) * N3_ + n];
    iw2p[t] = p.u;
  }
}

// ------------- per-row prep: nearest index, rel, imnet layer1 (gelu), score MLP
// One wave per row; lane computes 8 of 256 hidden units.
__global__ __launch_bounds__(256) void itnsr_prep(
    const float* __restrict__ coord, const float* __restrict__ scale,
    const float* __restrict__ iw0, const float* __restrict__ ib0,
    const float* __restrict__ sw0, const float* __restrict__ sb0,
    const float* __restrict__ sw1, const float* __restrict__ sb1,
    _Float16* __restrict__ h1, float* __restrict__ score,
    int* __restrict__ pixidx)
{
  const int tid = threadIdx.x, wave = tid >> 5, lane = tid & 31;
  const int r = blockIdx.x * 8 + wave;               // 0..65535
  const int s = r >> 14, b = (r >> 13) & 1, q = r & 8191;
  const float vx = (s & 2) ? 1.f : -1.f;
  const float vy = (s & 1) ? 1.f : -1.f;
  const size_t cq = ((size_t)b * Q_ + q) * 2;
  const float cx = coord[cq + 0], cy = coord[cq + 1];
  const float sx = scale[cq + 0], sy = scale[cq + 1];
  const float s00 = scale[(size_t)b * Q_ * 2 + 0];
  const float s01 = scale[(size_t)b * Q_ * 2 + 1];
  const float rx = (1.0f - s00) / 63.0f;             // 1/tx
  const float ry = (1.0f - s01) / 63.0f;
  float px = fminf(fmaxf(cx + vx * rx + 1e-6f, -1.f + 1e-6f), 1.f - 1e-6f);
  float py = fminf(fmaxf(cy + vy * ry + 1e-6f, -1.f + 1e-6f), 1.f - 1e-6f);
  int ih = (int)fminf(fmaxf(rintf((px + 1.f) * 32.f - 0.5f), 0.f), 63.f);
  int iw = (int)fminf(fmaxf(rintf((py + 1.f) * 32.f - 0.5f), 0.f), 63.f);
  float ckx = -1.f + (2.f * (float)ih + 1.f) / 64.f;
  float cky = -1.f + (2.f * (float)iw + 1.f) / 64.f;
  float relx = (cx - ckx) * 64.f;
  float rely = (cy - cky) * 64.f;
  float stx = sx * 64.f, sty = sy * 64.f;
  if (lane == 0) pixidx[r] = (ih << 6) | iw;
  float sacc = 0.f;
  #pragma unroll
  for (int jj = 0; jj < 8; ++jj) {
    int j = lane * 8 + jj;
    float h = relx * iw0[j] + rely * iw0[256 + j] +
              stx * iw0[512 + j] + sty * iw0[768 + j] + ib0[j];
    h1[(size_t)r * HID_ + j] = (_Float16)gelu_tanh(h);
    float sh = relx * sw0[j] + rely * sw0[256 + j] + sb0[j];
    sacc += gelu_tanh(sh) * sw1[j];
  }
  #pragma unroll
  for (int off = 16; off >= 1; off >>= 1) sacc += __shfl_xor(sacc, off, 32);
  if (lane == 0) score[r] = sacc + sb1[0];
}

// -------------------------- gemm2: h2 = gelu(h1 @ iw1 + ib1), WMMA f16 -> f32
__global__ __launch_bounds__(256) void itnsr_gemm2(
    const _Float16* __restrict__ h1, const uint32_t* __restrict__ iw1p,
    const float* __restrict__ ib1, _Float16* __restrict__ h2)
{
  __shared__ uint32_t AshU[16 * 128];    // 16 rows x 256 halves (8 KB)
  __shared__ uint32_t Bsh[64 * 128];     // pair-packed K-chunk (32 KB)
  const int tid = threadIdx.x, wave = tid >> 5, lane = tid & 31;
  const int r0 = blockIdx.x << 4;
  const int n0 = blockIdx.y << 7;
  const uint32_t* h1u = (const uint32_t*)(h1 + (size_t)r0 * HID_);
  for (int i = tid; i < 16 * 128; i += 256) AshU[i] = h1u[i];
  const int hi = (lane >= 16);
  const int sub = lane & 15;
  const int ncol = (wave << 4) + sub;
  v8f acc = {0.f, 0.f, 0.f, 0.f, 0.f, 0.f, 0.f, 0.f};
  for (int kc = 0; kc < 2; ++kc) {
    __syncthreads();
    for (int i = tid; i < 64 * 128; i += 256) {
      int k2 = i >> 7, n = i & 127;
      Bsh[i] = iw1p[(size_t)(kc * 64 + k2) * HID_ + (n0 + n)];
    }
    __syncthreads();
    #pragma unroll
    for (int ks = 0; ks < 4; ++ks) {
      const int kb = kc * 128 + ks * 32;
      AFrag a, bf;
      #pragma unroll
      for (int v = 0; v < 8; ++v) {
        int kA = kb + (hi ? 16 : 0) + 2 * v;
        a.u[v]  = AshU[sub * 128 + (kA >> 1)];
        bf.u[v] = Bsh[(ks * 16 + (hi ? 8 : 0) + v) * 128 + ncol];
      }
      acc = __builtin_amdgcn_wmma_f32_16x16x32_f16(
          false, a.v, false, bf.v, (short)0, acc, false, false);
    }
  }
  const int n = n0 + ncol;
  const int mb = hi ? 8 : 0;
  #pragma unroll
  for (int i = 0; i < 8; ++i) {
    float x = acc[i] + ib1[n];
    h2[(size_t)(r0 + mb + i) * HID_ + n] = (_Float16)gelu_tanh(x);
  }
}

// --- gemm3 fused: t = h2 @ iw2 + ib2, pred[k] = sum_c value[c]*t[3c+k] + score
// 16 rows/block; 14 column tiles of 128; wmat never hits HBM. Deterministic
// per-wave LDS accumulators (no atomics).
__global__ __launch_bounds__(256) void itnsr_gemm3(
    const _Float16* __restrict__ h2, const uint32_t* __restrict__ iw2p,
    const float* __restrict__ ib2, const float* __restrict__ feat,
    const int* __restrict__ pixidx, const float* __restrict__ score,
    float* __restrict__ stacked)
{
  __shared__ uint32_t AshU[16 * 128];    // 8 KB
  __shared__ uint32_t Bsh[32 * 128];     // 16 KB, K-chunk of 64
  __shared__ _Float16 Vsh[16 * C9_];     // 18 KB gathered values
  __shared__ float    Tsh[8][16][16];    // 8 KB per-wave GEMM tile
  __shared__ float    Qsh[8][16][3];     // 1.5 KB per-wave accumulators
  const int tid = threadIdx.x, wave = tid >> 5, lane = tid & 31;
  const int r0 = blockIdx.x << 4;
  const uint32_t* h2u = (const uint32_t*)(h2 + (size_t)r0 * HID_);
  for (int i = tid; i < 16 * 128; i += 256) AshU[i] = h2u[i];
  for (int i = tid; i < 16 * C9_; i += 256) {
    int m = i / C9_, cc = i - m * C9_;
    int c = cc / 9, k = cc - c * 9;
    int r = r0 + m;
    int bb = (r >> 13) & 1;
    int pix = pixidx[r];
    int y = (pix >> 6) + (k / 3) - 1;
    int x = (pix & 63) + (k % 3) - 1;
    float v = 0.f;
    if ((unsigned)y < 64u && (unsigned)x < 64u)
      v = feat[(((size_t)bb * C_ + c) << 12) + (y << 6) + x];
    Vsh[i] = (_Float16)v;
  }
  for (int i = tid; i < 8 * 16 * 3; i += 256) ((float*)Qsh)[i] = 0.f;
  const int hi = (lane >= 16);
  const int sub = lane & 15;
  const int ncol = (wave << 4) + sub;
  for (int nt = 0; nt < 14; ++nt) {
    const int colbase = nt << 7;
    if (nt < 13) __builtin_prefetch(&iw2p[(size_t)(colbase + 128)], 0, 1);
    v8f acc = {0.f, 0.f, 0.f, 0.f, 0.f, 0.f, 0.f, 0.f};
    for (int kc = 0; kc < 4; ++kc) {
      __syncthreads();
      for (int i = tid; i < 32 * 128; i += 256) {
        int k2 = i >> 7, n = i & 127;
        int col = colbase + n;
        Bsh[i] = (col < N3_) ? iw2p[(size_t)(kc * 32 + k2) * N3_ + col] : 0u;
      }
      __syncthreads();
      #pragma unroll
      for (int ks = 0; ks < 2; ++ks) {
        const int kb = kc * 64 + ks * 32;
        AFrag a, bf;
        #pragma unroll
        for (int v = 0; v < 8; ++v) {
          int kA = kb + (hi ? 16 : 0) + 2 * v;
          a.u[v]  = AshU[sub * 128 + (kA >> 1)];
          bf.u[v] = Bsh[(ks * 16 + (hi ? 8 : 0) + v) * 128 + ncol];
        }
        acc = __builtin_amdgcn_wmma_f32_16x16x32_f16(
            false, a.v, false, bf.v, (short)0, acc, false, false);
      }
    }
    const int mb = hi ? 8 : 0;
    #pragma unroll
    for (int i = 0; i < 8; ++i) Tsh[wave][mb + i][sub] = acc[i];
    __syncthreads();
    // Ordered (deterministic) value-contraction into per-wave accumulators.
    for (int i = tid; i < 384; i += 256) {
      int w = i / 48, rem = i - w * 48;
      int m = rem / 3, k3 = rem - m * 3;
      int cb = colbase + (w << 4);
      float a = 0.f;
      for (int n = 0; n < 16; ++n) {
        int col = cb + n;
        if (col < N3_ && (col % 3) == k3)
          a += (float)Vsh[m * C9_ + col / 3] * (Tsh[w][m][n] + ib2[col]);
      }
      Qsh[w][m][k3] += a;
    }
  }
  __syncthreads();
  if (tid < 48) {
    int m = tid / 3, k3 = tid - m * 3;
    int r = r0 + m;
    float acc = score[r];
    #pragma unroll
    for (int w = 0; w < 8; ++w) acc += Qsh[w][m][k3];
    int s = r >> 14, bb = (r >> 13) & 1, q = r & 8191;
    stacked[(((size_t)bb * Q_ + q) * 3 + k3) * 4 + s] = acc;
  }
}

// ------------------------------- final weight MLP 4->256(gelu)->1 per (b,q,k)
__global__ __launch_bounds__(256) void itnsr_wmlp(
    const float* __restrict__ stacked, const float* __restrict__ ww0,
    const float* __restrict__ wb0, const float* __restrict__ ww1,
    const float* __restrict__ wb1, float* __restrict__ out)
{
  const int tid = threadIdx.x, wave = tid >> 5, lane = tid & 31;
  const int row = blockIdx.x * 8 + wave;             // 0..49151
  const float x0 = stacked[(size_t)row * 4 + 0];
  const float x1 = stacked[(size_t)row * 4 + 1];
  const float x2 = stacked[(size_t)row * 4 + 2];
  const float x3 = stacked[(size_t)row * 4 + 3];
  float acc = 0.f;
  #pragma unroll
  for (int jj = 0; jj < 8; ++jj) {
    int j = lane * 8 + jj;
    float h = x0 * ww0[j] + x1 * ww0[256 + j] +
              x2 * ww0[512 + j] + x3 * ww0[768 + j] + wb0[j];
    acc += gelu_tanh(h) * ww1[j];
  }
  #pragma unroll
  for (int off = 16; off >= 1; off >>= 1) acc += __shfl_xor(acc, off, 32);
  if (lane == 0) out[row] = acc + wb1[0];
}

// ----------------------------------------------------------------------- host
extern "C" void kernel_launch(void* const* d_in, const int* in_sizes, int n_in,
                              void* d_out, int out_size, void* d_ws, size_t ws_size,
                              hipStream_t stream)
{
  (void)in_sizes; (void)n_in; (void)out_size; (void)ws_size;
  const float* inp   = (const float*)d_in[0];
  const float* coord = (const float*)d_in[1];
  const float* scale = (const float*)d_in[2];
  const float* enc_w = (const float*)d_in[3];
  const float* enc_b = (const float*)d_in[4];
  const float* iw0   = (const float*)d_in[5];
  const float* ib0   = (const float*)d_in[6];
  const float* iw1   = (const float*)d_in[7];
  const float* ib1   = (const float*)d_in[8];
  const float* iw2   = (const float*)d_in[9];
  const float* ib2   = (const float*)d_in[10];
  const float* sw0   = (const float*)d_in[11];
  const float* sb0   = (const float*)d_in[12];
  const float* sw1   = (const float*)d_in[13];
  const float* sb1   = (const float*)d_in[14];
  const float* ww0   = (const float*)d_in[15];
  const float* wb0   = (const float*)d_in[16];
  const float* ww1   = (const float*)d_in[17];
  const float* wb1   = (const float*)d_in[18];

  char* ws = (char*)d_ws;
  constexpr size_t off_feat  = 0;                                        // 2 MB
  constexpr size_t off_iw1p  = off_feat  + (size_t)B_ * C_ * H_ * W_ * 4;
  constexpr size_t off_iw2p  = off_iw1p  + (size_t)128 * HID_ * 4;       // 128 KB
  constexpr size_t off_h1    = off_iw2p  + (size_t)128 * N3_ * 4;        // 864 KB
  constexpr size_t off_h2    = off_h1    + (size_t)NROWS_ * HID_ * 2;    // 32 MB
  constexpr size_t off_score = off_h2    + (size_t)NROWS_ * HID_ * 2;    // 32 MB
  constexpr size_t off_pix   = off_score + (size_t)NROWS_ * 4;
  constexpr size_t off_stack = off_pix   + (size_t)NROWS_ * 4;
  // total ~= 72 MB of workspace

  float*     feat   = (float*)(ws + off_feat);
  uint32_t*  iw1p   = (uint32_t*)(ws + off_iw1p);
  uint32_t*  iw2p   = (uint32_t*)(ws + off_iw2p);
  _Float16*  h1     = (_Float16*)(ws + off_h1);
  _Float16*  h2     = (_Float16*)(ws + off_h2);
  float*     score  = (float*)(ws + off_score);
  int*       pixidx = (int*)(ws + off_pix);
  float*     stack  = (float*)(ws + off_stack);
  float*     out    = (float*)d_out;

  itnsr_enc  <<<2048, 256, 0, stream>>>(inp, enc_w, enc_b, feat);
  itnsr_packw<<< 992, 256, 0, stream>>>(iw1, iw2, iw1p, iw2p);
  itnsr_prep <<<8192, 256, 0, stream>>>(coord, scale, iw0, ib0, sw0, sb0,
                                        sw1, sb1, h1, score, pixidx);
  itnsr_gemm2<<<dim3(4096, 2), 256, 0, stream>>>(h1, iw1p, ib1, h2);
  itnsr_gemm3<<<4096, 256, 0, stream>>>(h2, iw2p, ib2, feat, pixidx, score, stack);
  itnsr_wmlp <<<6144, 256, 0, stream>>>(stack, ww0, wb0, ww1, wb1, out);
}
